// PerceiverARModel_56513179680777
// MI455X (gfx1250) — matrix operand order repaired and checked by reading, not compiled, not run on hardware
//
#include <hip/hip_runtime.h>

typedef __attribute__((ext_vector_type(16))) _Float16 v16h;
typedef __attribute__((ext_vector_type(8)))  float    v8f;

union H16 { v16h h; uint32_t u[8]; };
union F8  { v8f  v; float    f[8]; };

#define WMMA_F16(A, B, C) \
  __builtin_amdgcn_wmma_f32_16x16x32_f16(false, (A), false, (B), (short)0, (C), false, false)

__device__ __forceinline__ void wave_lds_fence() {
  // wave-lockstep cross-lane LDS visibility: drain DS counter
  asm volatile("s_wait_dscnt 0" ::: "memory");
}
__device__ __forceinline__ void wait_async() {
  asm volatile("s_wait_asynccnt 0" ::: "memory");
}
// CDNA5 async copy: global -> LDS, 128b per lane, tracked by ASYNCcnt.
// LDS dest address = low 32 bits of the generic shared pointer (ISA 10.2).
__device__ __forceinline__ void async_load_b128(void* lds_dst, const void* gaddr) {
  unsigned off = (unsigned)(uintptr_t)lds_dst;
  asm volatile("global_load_async_to_lds_b128 %0, %1, off"
               :: "v"(off), "v"(gaddr) : "memory");
}

// ---------------------------------------------------------------------------
// f32 -> f16 conversion (weights)
// ---------------------------------------------------------------------------
__global__ void cvt_f16(const float* __restrict__ in, _Float16* __restrict__ out, int n) {
  int i = blockIdx.x * blockDim.x + threadIdx.x;
  if (i < n) out[i] = (_Float16)in[i];
}

// ---------------------------------------------------------------------------
// LayerNorm over D=512, fp32 in -> f16 out. One row per 256-thread block.
// ---------------------------------------------------------------------------
__global__ __launch_bounds__(256) void layernorm_f16(const float* __restrict__ X,
                                                     const float* __restrict__ g,
                                                     const float* __restrict__ bta,
                                                     _Float16* __restrict__ Y) {
  const int row = blockIdx.x;
  const int tid = threadIdx.x;
  const float* x = X + (size_t)row * 512;
  __shared__ float s1[256], s2[256];
  float a = x[tid], b = x[tid + 256];
  s1[tid] = a + b;
  s2[tid] = a * a + b * b;
  __syncthreads();
  for (int st = 128; st > 0; st >>= 1) {
    if (tid < st) { s1[tid] += s1[tid + st]; s2[tid] += s2[tid + st]; }
    __syncthreads();
  }
  float mean = s1[0] * (1.0f / 512.0f);
  float var  = s2[0] * (1.0f / 512.0f) - mean * mean;
  float rstd = rsqrtf(var + 1e-5f);
  Y[(size_t)row * 512 + tid]       = (_Float16)((a - mean) * rstd * g[tid] + bta[tid]);
  Y[(size_t)row * 512 + tid + 256] = (_Float16)((b - mean) * rstd * g[tid + 256] + bta[tid + 256]);
}

// ---------------------------------------------------------------------------
// WMMA GEMM: C[Mr,Nc] = A[Mr,K](f16) x B[K,Nc](f16), fused templated epilogue.
// Block tile 128x64, 8 waves (4x2), each wave 32x32 (2x2 WMMA tiles), K-step 32.
// Requires Mr%128==0, Nc%64==0, K%32==0 (true for all uses here).
// EPI: 0 = f16 out, 1 = f16 out + exact GELU, 2 = f32 out + bias + resid,
//      3 = f32 out + resid.
// REMAP (f16 out only): pr = baserow + r + (r/seg)*jump   (KV interleave).
// ---------------------------------------------------------------------------
template <int EPI, bool REMAP>
__global__ __launch_bounds__(256) void gemm_wmma(
    const _Float16* __restrict__ A, int lda,
    const _Float16* __restrict__ Bw, int ldb,   // ldb == Nc
    int K,
    float* __restrict__ out32,
    _Float16* __restrict__ out16,
    int ldo,
    const float* __restrict__ bias,
    const float* __restrict__ resid,
    int seg, int jump, int baserow) {
  __shared__ _Float16 As[128][40];   // [m][k], pad keeps 16B alignment
  __shared__ _Float16 Bs[64][40];    // [n][k] (transposed)

  const int tid  = threadIdx.x;
  const int lane = tid & 31, wave = tid >> 5;
  const int wm = wave & 3, wn = wave >> 2;
  const int half = lane >> 4, l16 = lane & 15;
  const int rowBase = blockIdx.x * 128;
  const int colBase = blockIdx.y * 64;

  v8f zero = {};
  F8 acc[2][2];
  acc[0][0].v = zero; acc[0][1].v = zero; acc[1][0].v = zero; acc[1][1].v = zero;

  const int nB  = tid & 63;
  const int kkB = (tid >> 6) << 3;

  for (int k0 = 0; k0 < K; k0 += 32) {
    __syncthreads();
    {  // A tile: 512 chunks of 8 f16, 2 per thread, async global->LDS (ASYNCcnt)
      int c0 = tid * 2;
      int r0 = c0 >> 2, cc0 = (c0 & 3) << 3;
      async_load_b128(&As[r0][cc0], A + (size_t)(rowBase + r0) * lda + (k0 + cc0));
      int c1 = c0 + 1;
      int r1 = c1 >> 2, cc1 = (c1 & 3) << 3;
      async_load_b128(&As[r1][cc1], A + (size_t)(rowBase + r1) * lda + (k0 + cc1));
    }
    {  // B tile transposed into [n][k] (manual path, overlaps the async copies)
#pragma unroll
      for (int i = 0; i < 4; ++i) {
        union { _Float16 h[2]; uint32_t u; } pk;
        pk.h[0] = Bw[(size_t)(k0 + kkB + 2 * i)     * ldb + colBase + nB];
        pk.h[1] = Bw[(size_t)(k0 + kkB + 2 * i + 1) * ldb + colBase + nB];
        *(uint32_t*)&Bs[nB][kkB + 2 * i] = pk.u;
      }
    }
    if (k0 + 32 < K) {  // prefetch next tiles into cache
      __builtin_prefetch(A + (size_t)(rowBase + (tid >> 1)) * lda + (k0 + 32));
      __builtin_prefetch(Bw + (size_t)(k0 + 32 + (tid >> 3)) * ldb + colBase);
    }
    wait_async();
    __syncthreads();

    H16 af[2], bf[2];
#pragma unroll
    for (int mt = 0; mt < 2; ++mt) {
      int m = wm * 32 + mt * 16 + l16;
#pragma unroll
      for (int v = 0; v < 8; ++v) {
        int k = ((v & 4) << 2) + half * 8 + ((v & 3) << 1);  // A-frag layout
        af[mt].u[v] = *(const uint32_t*)&As[m][k];
      }
    }
#pragma unroll
    for (int nt = 0; nt < 2; ++nt) {
      int n = wn * 32 + nt * 16 + l16;
#pragma unroll
      for (int v = 0; v < 8; ++v) {
        int k = ((v & 4) << 2) + half * 8 + ((v & 3) << 1);  // B-frag layout
        bf[nt].u[v] = *(const uint32_t*)&Bs[n][k];
      }
    }
    acc[0][0].v = WMMA_F16(af[0].h, bf[0].h, acc[0][0].v);
    acc[0][1].v = WMMA_F16(af[0].h, bf[1].h, acc[0][1].v);
    acc[1][0].v = WMMA_F16(af[1].h, bf[0].h, acc[1][0].v);
    acc[1][1].v = WMMA_F16(af[1].h, bf[1].h, acc[1][1].v);
  }

#pragma unroll
  for (int mt = 0; mt < 2; ++mt) {
#pragma unroll
    for (int nt = 0; nt < 2; ++nt) {
      int Cc = colBase + wn * 32 + nt * 16 + l16;
      float bcol = (EPI == 2) ? bias[Cc] : 0.0f;
#pragma unroll
      for (int r = 0; r < 8; ++r) {
        int Rr = rowBase + wm * 32 + mt * 16 + half * 8 + r;  // C-frag layout
        float val = acc[mt][nt].f[r];
        if (EPI == 2) val += bcol + resid[(size_t)Rr * ldo + Cc];
        if (EPI == 3) val += resid[(size_t)Rr * ldo + Cc];
        if (EPI == 1) val = 0.5f * val * (1.0f + erff(val * 0.70710678118654752f));
        if (EPI <= 1) {
          int pr = REMAP ? (baserow + Rr + (Rr / seg) * jump) : Rr;
          out16[(size_t)pr * ldo + Cc] = (_Float16)val;
        } else {
          out32[(size_t)Rr * ldo + Cc] = val;
        }
      }
    }
  }
}

// ---------------------------------------------------------------------------
// Flash attention (f16 in, f16 out, fp32 accum + online softmax).
// Grid: (Nq/64, B*H); 128 threads = 4 waves x 16 query rows; key chunks of 32.
// Mask: key j allowed iff j <= i + offs.
// ---------------------------------------------------------------------------
__global__ __launch_bounds__(128) void attn_kernel(
    const _Float16* __restrict__ Qp, int qld, int qoff,
    const _Float16* __restrict__ Kp, int kld, int koff,
    const _Float16* __restrict__ Vp, int vld, int voff,
    _Float16* __restrict__ Op, int oldd,
    int Nq, int J, int offs) {
  __shared__ _Float16 Ks[32][64];      // [key][d]  == B-frag order for QK^T
  __shared__ _Float16 Vt[64][40];      // [d][key]  == B-frag order for PV
  __shared__ float    Ssc[4][16][33];  // per-wave score staging
  __shared__ _Float16 Pb[4][16][32];   // per-wave probabilities
  __shared__ float    redA[4][16];     // per-row rescale alpha
  __shared__ float    redL[4][16];     // per-row 1/l

  const int b = blockIdx.y >> 3, h = blockIdx.y & 7;
  const int qbase = blockIdx.x * 64;
  const int tid  = threadIdx.x;
  const int lane = tid & 31, wave = tid >> 5;
  const int half = lane >> 4, l16 = lane & 15;
  const int qw = qbase + wave * 16;

  const _Float16* Qb = Qp + (size_t)b * Nq * qld + qoff + h * 64;
  const _Float16* Kb = Kp + (size_t)b * J  * kld + koff + h * 64;
  const _Float16* Vb = Vp + (size_t)b * J  * vld + voff + h * 64;

  // Q fragments (resident for whole key loop): A-frag layout, d = 0..63
  H16 aq[2];
  {
    int i = qw + l16;
#pragma unroll
    for (int kc = 0; kc < 2; ++kc)
#pragma unroll
      for (int v = 0; v < 8; ++v) {
        int d = kc * 32 + ((v & 4) << 2) + half * 8 + ((v & 3) << 1);
        aq[kc].u[v] = *(const uint32_t*)(Qb + (size_t)i * qld + d);
      }
  }

  v8f zero = {};
  F8 o[4];
  o[0].v = zero; o[1].v = zero; o[2].v = zero; o[3].v = zero;
  float mrow = -3.0e38f, lrow = 0.0f;

  const int jlimit = (J < qbase + 64 + offs) ? J : (qbase + 64 + offs);

  for (int j0 = 0; j0 < jlimit; j0 += 32) {
    __syncthreads();
    {  // K chunk: 32 keys x 64 d, async global->LDS (128b per lane)
      int c0 = tid * 2;
      int key0 = c0 >> 3, cc0 = (c0 & 7) << 3;
      async_load_b128(&Ks[key0][cc0], Kb + (size_t)(j0 + key0) * kld + cc0);
      int c1 = c0 + 1;
      int key1 = c1 >> 3, cc1 = (c1 & 7) << 3;
      async_load_b128(&Ks[key1][cc1], Kb + (size_t)(j0 + key1) * kld + cc1);
    }
    {  // V chunk transposed to [d][key] (manual, overlaps the async copies)
      int key = tid & 31, dg = (tid >> 5) * 16;
      _Float16 tv[16];
      *(int4*)&tv[0] = *(const int4*)(Vb + (size_t)(j0 + key) * vld + dg);
      *(int4*)&tv[8] = *(const int4*)(Vb + (size_t)(j0 + key) * vld + dg + 8);
#pragma unroll
      for (int ii = 0; ii < 16; ++ii) Vt[dg + ii][key] = tv[ii];
    }
    if (j0 + 32 < jlimit) {  // prefetch next K/V rows
      __builtin_prefetch(Kb + (size_t)(j0 + 32 + (tid >> 2)) * kld);
      __builtin_prefetch(Vb + (size_t)(j0 + 32 + (tid >> 2)) * vld);
    }
    wait_async();
    __syncthreads();

    // scores: two 16-key sub-tiles, each = 2 WMMAs over d
#pragma unroll
    for (int s = 0; s < 2; ++s) {
      H16 bk0, bk1;
#pragma unroll
      for (int v = 0; v < 8; ++v) {
        int d0 = ((v & 4) << 2) + half * 8 + ((v & 3) << 1);
        bk0.u[v] = *(const uint32_t*)&Ks[s * 16 + l16][d0];
        bk1.u[v] = *(const uint32_t*)&Ks[s * 16 + l16][32 + d0];
      }
      F8 sc; sc.v = zero;
      sc.v = WMMA_F16(aq[0].h, bk0.h, sc.v);
      sc.v = WMMA_F16(aq[1].h, bk1.h, sc.v);
#pragma unroll
      for (int r = 0; r < 8; ++r)
        Ssc[wave][half * 8 + r][s * 16 + l16] = sc.f[r] * 0.125f;  // * DH^-0.5
    }
    wave_lds_fence();

    // online softmax: lane = query row (lanes 0..15 of each wave)
    if (lane < 16) {
      int i = qw + lane;
      float sv[32];
      float mnew = mrow;
#pragma unroll
      for (int c = 0; c < 32; ++c) {
        float s = Ssc[wave][lane][c];
        if (j0 + c > i + offs) s = -3.0e38f;
        sv[c] = s;
        mnew = fmaxf(mnew, s);
      }
      float alpha = expf(mrow - mnew);
      float ls = 0.0f;
#pragma unroll
      for (int c = 0; c < 32; ++c) {
        float pv = (sv[c] <= -1.0e37f) ? 0.0f : expf(sv[c] - mnew);
        ls += pv;
        Pb[wave][lane][c] = (_Float16)pv;
      }
      lrow = lrow * alpha + ls;
      mrow = mnew;
      redA[wave][lane] = alpha;
    }
    wave_lds_fence();

    // rescale O accumulators by alpha (per-row broadcast via LDS)
    float ar[8];
#pragma unroll
    for (int r = 0; r < 8; ++r) ar[r] = redA[wave][half * 8 + r];
#pragma unroll
    for (int t = 0; t < 4; ++t)
#pragma unroll
      for (int r = 0; r < 8; ++r) o[t].f[r] *= ar[r];

    // P as A-fragment [16q x 32k], then O += P @ V (4 WMMAs across d)
    H16 ap;
#pragma unroll
    for (int v = 0; v < 8; ++v) {
      int k = ((v & 4) << 2) + half * 8 + ((v & 3) << 1);
      ap.u[v] = *(const uint32_t*)&Pb[wave][l16][k];
    }
#pragma unroll
    for (int t = 0; t < 4; ++t) {
      H16 bv;
#pragma unroll
      for (int v = 0; v < 8; ++v) {
        int k = ((v & 4) << 2) + half * 8 + ((v & 3) << 1);
        bv.u[v] = *(const uint32_t*)&Vt[t * 16 + l16][k];
      }
      o[t].v = WMMA_F16(ap.h, bv.h, o[t].v);
    }
  }

  if (lane < 16) redL[wave][lane] = 1.0f / lrow;
  wave_lds_fence();
  float lr[8];
#pragma unroll
  for (int r = 0; r < 8; ++r) lr[r] = redL[wave][half * 8 + r];

  _Float16* Ob = Op + (size_t)b * Nq * oldd + h * 64;
#pragma unroll
  for (int t = 0; t < 4; ++t)
#pragma unroll
    for (int r = 0; r < 8; ++r) {
      int i = qw + half * 8 + r;
      Ob[(size_t)i * oldd + t * 16 + l16] = (_Float16)(o[t].f[r] * lr[r]);
    }
}

// ---------------------------------------------------------------------------
// Host orchestration
// ---------------------------------------------------------------------------
extern "C" void kernel_launch(void* const* d_in, const int* in_sizes, int n_in,
                              void* d_out, int out_size, void* d_ws, size_t ws_size,
                              hipStream_t stream) {
  (void)in_sizes; (void)n_in; (void)out_size; (void)ws_size;

  const float* x       = (const float*)d_in[0];
  const float* context = (const float*)d_in[1];
  const float* pa_ng   = (const float*)d_in[2];
  const float* pa_nb   = (const float*)d_in[3];
  const float* pa_cng  = (const float*)d_in[4];
  const float* pa_cnb  = (const float*)d_in[5];
  const float* pa_wq   = (const float*)d_in[6];
  const float* pa_wkv  = (const float*)d_in[7];
  const float* pa_wo   = (const float*)d_in[8];
  const float* pa_wo_b = (const float*)d_in[9];
  const float* pf_g    = (const float*)d_in[10];
  const float* pf_b    = (const float*)d_in[11];
  const float* pf_w1   = (const float*)d_in[12];
  const float* pf_w2   = (const float*)d_in[13];
  const float* sa_g    = (const float*)d_in[14];
  const float* sa_b    = (const float*)d_in[15];
  const float* sa_wqkv = (const float*)d_in[16];
  const float* sa_wo   = (const float*)d_in[17];
  const float* sf_g    = (const float*)d_in[18];
  const float* sf_b    = (const float*)d_in[19];
  const float* sf_w1   = (const float*)d_in[20];
  const float* sf_w2   = (const float*)d_in[21];

  float* resid = (float*)d_out;  // residual stream [4096, 512] fp32

  char* p = (char*)d_ws;
  auto a16 = [&](size_t n) {
    _Float16* r = (_Float16*)p;
    p += ((n * sizeof(_Float16) + 255) & ~(size_t)255);
    return r;
  };

  _Float16* w_q    = a16((size_t)512 * 512);
  _Float16* w_kv   = a16((size_t)512 * 1024);
  _Float16* w_o    = a16((size_t)512 * 512);
  _Float16* w_p1   = a16((size_t)512 * 2048);
  _Float16* w_p2   = a16((size_t)2048 * 512);
  _Float16* w_sqkv = a16((size_t)6 * 512 * 1536);
  _Float16* w_so   = a16((size_t)6 * 512 * 512);
  _Float16* w_s1   = a16((size_t)6 * 512 * 2048);
  _Float16* w_s2   = a16((size_t)6 * 2048 * 512);
  _Float16* xn16   = a16((size_t)4096 * 512);
  _Float16* cn16   = a16((size_t)12288 * 512);
  _Float16* q16    = a16((size_t)4096 * 512);
  _Float16* kv16   = a16((size_t)4 * 4096 * 1024);
  _Float16* qkv16  = a16((size_t)4096 * 1536);
  _Float16* at16   = a16((size_t)4096 * 512);
  _Float16* ff16   = a16((size_t)4096 * 2048);

  auto cvt = [&](const float* src, _Float16* dst, size_t n) {
    cvt_f16<<<dim3((unsigned)((n + 255) / 256)), 256, 0, stream>>>(src, dst, (int)n);
  };
  cvt(pa_wq,   w_q,    (size_t)512 * 512);
  cvt(pa_wkv,  w_kv,   (size_t)512 * 1024);
  cvt(pa_wo,   w_o,    (size_t)512 * 512);
  cvt(pf_w1,   w_p1,   (size_t)512 * 2048);
  cvt(pf_w2,   w_p2,   (size_t)2048 * 512);
  cvt(sa_wqkv, w_sqkv, (size_t)6 * 512 * 1536);
  cvt(sa_wo,   w_so,   (size_t)6 * 512 * 512);
  cvt(sf_w1,   w_s1,   (size_t)6 * 512 * 2048);
  cvt(sf_w2,   w_s2,   (size_t)6 * 2048 * 512);

  hipMemcpyAsync(resid, x, (size_t)4096 * 512 * sizeof(float),
                 hipMemcpyDeviceToDevice, stream);

  auto ln = [&](const float* src, const float* g, const float* bb, _Float16* dst, int rows) {
    layernorm_f16<<<dim3(rows), 256, 0, stream>>>(src, g, bb, dst);
  };

  // EPI: 0 f16 out, 1 f16+gelu, 2 f32+bias+resid, 3 f32+resid
  auto gemm = [&](int epi, bool remap,
                  const _Float16* A, int lda, const _Float16* Bw, int Nc, int Kc, int Mr,
                  float* o32, _Float16* o16, const float* bias, const float* res,
                  int seg, int jump, int baser) {
    dim3 g(Mr / 128, Nc / 64);
    if (epi == 0 && !remap)
      gemm_wmma<0, false><<<g, 256, 0, stream>>>(A, lda, Bw, Nc, Kc, o32, o16, Nc, bias, res, seg, jump, baser);
    else if (epi == 0)
      gemm_wmma<0, true><<<g, 256, 0, stream>>>(A, lda, Bw, Nc, Kc, o32, o16, Nc, bias, res, seg, jump, baser);
    else if (epi == 1)
      gemm_wmma<1, false><<<g, 256, 0, stream>>>(A, lda, Bw, Nc, Kc, o32, o16, Nc, bias, res, seg, jump, baser);
    else if (epi == 2)
      gemm_wmma<2, false><<<g, 256, 0, stream>>>(A, lda, Bw, Nc, Kc, o32, o16, Nc, bias, res, seg, jump, baser);
    else
      gemm_wmma<3, false><<<g, 256, 0, stream>>>(A, lda, Bw, Nc, Kc, o32, o16, Nc, bias, res, seg, jump, baser);
  };

  // ---- CausalPrefixAttention ----
  ln(resid, pa_ng, pa_nb, xn16, 4096);
  ln(context, pa_cng, pa_cnb, cn16, 12288);
  gemm(0, false, xn16, 512, w_q, 512, 512, 4096, nullptr, q16, nullptr, nullptr, 0, 0, 0);
  // kv interleave: rows b*3072+m -> b*4096+m ; rows b*1024+i -> b*4096+3072+i
  gemm(0, true, cn16, 512, w_kv, 1024, 512, 12288, nullptr, kv16, nullptr, nullptr, 3072, 1024, 0);
  gemm(0, true, xn16, 512, w_kv, 1024, 512, 4096,  nullptr, kv16, nullptr, nullptr, 1024, 3072, 3072);
  attn_kernel<<<dim3(16, 32), 128, 0, stream>>>(q16, 512, 0, kv16, 1024, 0, kv16, 1024, 512,
                                                at16, 512, 1024, 4096, 3072);
  gemm(2, false, at16, 512, w_o, 512, 512, 4096, resid, nullptr, pa_wo_b, resid, 0, 0, 0);
  // prefix FF
  ln(resid, pf_g, pf_b, xn16, 4096);
  gemm(1, false, xn16, 512, w_p1, 2048, 512, 4096, nullptr, ff16, nullptr, nullptr, 0, 0, 0);
  gemm(3, false, ff16, 2048, w_p2, 512, 2048, 4096, resid, nullptr, nullptr, resid, 0, 0, 0);

  // ---- depth x (CausalAttention + FeedForward) ----
  for (int l = 0; l < 6; ++l) {
    ln(resid, sa_g + (size_t)l * 512, sa_b + (size_t)l * 512, xn16, 4096);
    gemm(0, false, xn16, 512, w_sqkv + (size_t)l * 512 * 1536, 1536, 512, 4096,
         nullptr, qkv16, nullptr, nullptr, 0, 0, 0);
    attn_kernel<<<dim3(16, 32), 128, 0, stream>>>(qkv16, 1536, 0, qkv16, 1536, 512,
                                                  qkv16, 1536, 1024, at16, 512,
                                                  1024, 1024, 0);
    gemm(3, false, at16, 512, w_so + (size_t)l * 512 * 512, 512, 512, 4096,
         resid, nullptr, nullptr, resid, 0, 0, 0);
    ln(resid, sf_g + (size_t)l * 512, sf_b + (size_t)l * 512, xn16, 4096);
    gemm(1, false, xn16, 512, w_s1 + (size_t)l * 512 * 2048, 2048, 512, 4096,
         nullptr, ff16, nullptr, nullptr, 0, 0, 0);
    gemm(3, false, ff16, 2048, w_s2 + (size_t)l * 2048 * 512, 512, 2048, 4096,
         resid, nullptr, nullptr, resid, 0, 0, 0);
  }
}
